// WaveNet_49185965474189
// MI455X (gfx1250) — compile-verified, matrix operand order checked by reading, not run on hardware
//
#include <hip/hip_runtime.h>
#include <math.h>

// ---------------- model constants ----------------
#define BN   2
#define SLEN 19200
#define HCH  64
#define NL   24
#define SKIPC 256
#define ACH  256

typedef _Float16 h16;
typedef __attribute__((ext_vector_type(16))) _Float16 v16h;
typedef __attribute__((ext_vector_type(2)))  _Float16 v2h;
typedef __attribute__((ext_vector_type(8)))  float    v8f;
typedef __attribute__((ext_vector_type(4)))  unsigned int v4u;
typedef __attribute__((ext_vector_type(8)))  int      v8i;
typedef __attribute__((ext_vector_type(4)))  int      v4i;

// ---------------- packed-weight region offsets (in halves) ----------------
// dilate: NL * 8 rowtiles * 4 ksteps * 512
// skip:   NL * 16 rowtiles * 2 ksteps * 512
// res:    (NL-1) * 4 rowtiles * 2 ksteps * 512
// out1/out2: 16 rowtiles * 8 ksteps * 512
#define N_DIL   (24*8*4*512)
#define OFF_SKP (N_DIL)
#define N_SKP   (24*16*2*512)
#define OFF_RES (OFF_SKP + N_SKP)
#define N_RES   (23*4*2*512)
#define OFF_O1  (OFF_RES + N_RES)
#define N_O1    (16*8*512)
#define OFF_O2  (OFF_O1 + N_O1)
#define TOTAL_PACK (OFF_O2 + N_O1)

// ---------------- WMMA helpers ----------------
__device__ __forceinline__ v8f wmma32(v16h a, v16h b, v8f c) {
  // 8 args: (neg_a, A, neg_b, B, c_mod, C, reuse_a, reuse_b)
  return __builtin_amdgcn_wmma_f32_16x16x32_f16(false, a, false, b, (short)0, c, false, false);
}

// B-fragment gather from LDS: matrix stored [col][row] (row contiguous, pitch halves).
// 32x16 B layout: lanes 0-15 = N, VGPR v holds K pair {pat(v), pat(v)+1}, +8 for lanes 16-31.
__device__ __forceinline__ v16h gatherB(const h16* sm, int pitch, int tc, int kb) {
  int lane = threadIdx.x & 31;
  int n = lane & 15, hs = lane >> 4;
  const h16* p = sm + (tc + n) * pitch + kb + hs * 8;
  v16h r;
#pragma unroll
  for (int v = 0; v < 8; ++v) {
    v2h x = *(const v2h*)(p + ((v < 4) ? 2 * v : 16 + 2 * (v - 4)));
    r[2 * v] = x.x; r[2 * v + 1] = x.y;
  }
  return r;
}

__device__ __forceinline__ v16h loadA(const h16* packW, int block) {
  int lane = threadIdx.x & 31;
  return *(const v16h*)(packW + (size_t)block * 512 + lane * 16);
}

// ---------------- weight packing ----------------
__global__ void pack_weights(const float* __restrict__ dw, const float* __restrict__ sw,
                             const float* __restrict__ rw, const float* __restrict__ o1,
                             const float* __restrict__ o2, h16* __restrict__ packW) {
  int idx = blockIdx.x * 256 + threadIdx.x;
  if (idx >= TOTAL_PACK) return;
  int lane = 0, e = 0, m = 0, hs = 0, k32 = 0;
  auto dec = [&](int relidx, int& block) {
    block = relidx / 512; int w = relidx % 512;
    lane = w / 16; e = w % 16; m = lane & 15; hs = lane >> 4;
    int v = e >> 1, h = e & 1;
    k32 = ((v < 4) ? 2 * v : 16 + 2 * (v - 4)) + h + 8 * hs;
  };
  float val;
  if (idx < OFF_SKP) {                       // dilate [NL][2H=128][64][2]; K = {tap0 ch, tap1 ch}
    int block; dec(idx, block);
    int s = block & 3, rt = (block >> 2) & 7, l = block >> 5;
    int row = rt * 16 + m, kg = s * 32 + k32, tap = kg >> 6, c = kg & 63;
    val = dw[((l * 128 + row) * 64 + c) * 2 + tap];
  } else if (idx < OFF_RES) {                // skip [NL][256][64]
    int block; dec(idx - OFF_SKP, block);
    int s = block & 1, rt = (block >> 1) & 15, l = block >> 5;
    val = sw[(l * 256 + rt * 16 + m) * 64 + s * 32 + k32];
  } else if (idx < OFF_O1) {                 // res [NL-1][64][64]
    int block; dec(idx - OFF_RES, block);
    int s = block & 1, rt = (block >> 1) & 3, l = block >> 3;
    val = rw[(l * 64 + rt * 16 + m) * 64 + s * 32 + k32];
  } else if (idx < OFF_O2) {                 // out1 [256][256]
    int block; dec(idx - OFF_O1, block);
    int s = block & 7, rt = block >> 3;
    val = o1[(rt * 16 + m) * 256 + s * 32 + k32];
  } else {                                    // out2 [256][256]
    int block; dec(idx - OFF_O2, block);
    int s = block & 7, rt = block >> 3;
    val = o2[(rt * 16 + m) * 256 + s * 32 + k32];
  }
  packW[idx] = (h16)val;
}

// ---------------- conditioning (tiny, VALU) ----------------
__global__ void up_kernel(const float* __restrict__ lf, const float* __restrict__ tw,
                          const float* __restrict__ tb, float* __restrict__ up) {
  int idx = blockIdx.x * 256 + threadIdx.x;
  if (idx >= BN * 80 * 256) return;
  int tt = idx & 255, f = (idx >> 8) % 80, b = idx / (80 * 256);
  int i = tt >> 2, k = tt & 3;
  float s = tb[f];
  for (int c = 0; c < 80; ++c)
    s += tw[(c * 80 + f) * 4 + k] * lf[(b * 64 + i) * 80 + c];
  up[idx] = s;                               // [b][f][tt]
}

__global__ void cond_kernel(const float* __restrict__ cw, const float* __restrict__ cb,
                            const float* __restrict__ up, float* __restrict__ cond256) {
  int idx = blockIdx.x * 256 + threadIdx.x;
  if (idx >= BN * NL * 128 * 256) return;
  int b = idx / (NL * 128 * 256), rem = idx % (NL * 128 * 256);
  int l = rem / (128 * 256), row = (rem / 256) & 127, tt = rem & 255;
  int o = l * 128 + row;
  float s = cb[o];
  for (int f = 0; f < 80; ++f)
    s += cw[o * 80 + f] * up[(b * 80 + f) * 256 + tt];
  cond256[idx] = s;                          // [b][l][row][tt]
}

__global__ void embed_kernel(const float* __restrict__ gold, const float* __restrict__ ew,
                             const float* __restrict__ eb, h16* __restrict__ x0) {
  int idx = blockIdx.x * 256 + threadIdx.x;
  if (idx >= BN * SLEN * 64) return;
  int c = idx & 63, t = (idx >> 6) % SLEN, b = idx / (SLEN * 64);
  x0[idx] = (h16)(gold[b * SLEN + t] * ew[c] + eb[c]);   // [b][t][c]
}

__global__ void zero_kernel(float4* __restrict__ p, int n4) {
  int i = blockIdx.x * 256 + threadIdx.x;
  if (i < n4) p[i] = make_float4(0.f, 0.f, 0.f, 0.f);
}

// ---------------- residual-stack layer ----------------
#define XPITCH 66   // 64 ch + 2 pad halves (1 dword) -> 33-dword row stride, conflict-free
#define YPITCH 132
#define GPITCH 66

__global__ __launch_bounds__(256) void layer_kernel(
    int l, int d,
    const h16* __restrict__ xin, h16* __restrict__ xout,
    const h16* __restrict__ packW, const float* __restrict__ cond256,
    const float* __restrict__ dil_b, const float* __restrict__ skp_b,
    const float* __restrict__ rs_b, float* __restrict__ skip) {
  __shared__ h16   Xsm[96 * XPITCH];   // t0-32 .. t0+63, [col][ch]
  __shared__ float Ysm[64 * YPITCH];   // [t][row 0..127]
  __shared__ h16   Gsm[64 * GPITCH];   // [t][ch 0..63]
  const int tid = threadIdx.x, wave = tid >> 5, lane = tid & 31;
  const int b = blockIdx.x / (SLEN / 64);
  const int t0 = (blockIdx.x % (SLEN / 64)) * 64;
  const int n = lane & 15, hs = lane >> 4;

  // ---- stage 0: x tile -> LDS (TDM when available) ----
  const int rows_lo = (t0 >= 32) ? 0 : (32 - t0);  // zero-pad causal left edge
  for (int i = tid; i < rows_lo * 64; i += 256) {
    int j = i >> 6, c = i & 63;
    Xsm[j * XPITCH + c] = (h16)0;
  }
#if __has_builtin(__builtin_amdgcn_tensor_load_to_lds)
  if (tid < 32) {  // one wave issues the tensor DMA
    int tstart = t0 - 32 + rows_lo;
    unsigned long long ga =
        (unsigned long long)(uintptr_t)(xin + ((size_t)b * SLEN + tstart) * 64);
    unsigned int ldsb = (unsigned int)(uintptr_t)(void*)Xsm + (unsigned int)(rows_lo * XPITCH * 2);
    v4u g0;
    g0.x = 1u;                                   // count=1, user descriptor
    g0.y = ldsb;                                 // lds_addr
    g0.z = (unsigned int)ga;                     // global_addr[31:0]
    g0.w = (unsigned int)(ga >> 32) | (2u << 30);// global_addr[56:32] | type=2
    int td1 = SLEN - tstart;
    v8i g1;
    g1[0] = (int)((1u << 16) | (1u << 20) | (4u << 22) | (0u << 25)); // 2B elems, pad_en, 32dw interval, 1dw pad
    g1[1] = (int)(64u << 16);                    // tensor_dim0 = 64 (low 16 in high half)
    g1[2] = (int)(((unsigned)td1 & 0xFFFFu) << 16);          // tensor_dim1 lo16
    g1[3] = (int)(((unsigned)td1 >> 16) | (64u << 16));      // tensor_dim1 hi | tile_dim0=64
    g1[4] = (int)(unsigned)(96 - rows_lo);       // tile_dim1 (rows), tile_dim2=0
    g1[5] = 64;                                  // tensor_dim0_stride
    g1[6] = 0; g1[7] = 0;
    v4i gz4 = {0, 0, 0, 0};
    v8i gz8 = {0, 0, 0, 0, 0, 0, 0, 0};
    __builtin_amdgcn_tensor_load_to_lds(g0, g1, gz4, gz4, gz8, 0);
    __builtin_amdgcn_s_wait_tensorcnt(0);
  }
#else
  for (int i = tid + rows_lo * 64; i < 96 * 64; i += 256) {
    int j = i >> 6, c = i & 63;
    int tg = t0 - 32 + j;
    Xsm[j * XPITCH + c] = xin[((size_t)b * SLEN + tg) * 64 + c];
  }
#endif
  __syncthreads();

  // ---- stage 1: dilated conv GEMM  y[128 x 64] = W[128 x 128] * [x(t-d); x(t)] ----
  {
    int rt = wave;
    int base = (l * 8 + rt) * 4;
    v16h A0 = loadA(packW, base + 0);
    v16h A1 = loadA(packW, base + 1);
    v16h A2 = loadA(packW, base + 2);
    v16h A3 = loadA(packW, base + 3);
#pragma unroll
    for (int ct = 0; ct < 4; ++ct) {
      v8f acc = {0.f, 0.f, 0.f, 0.f, 0.f, 0.f, 0.f, 0.f};
      int tc = 32 + ct * 16;
      acc = wmma32(A0, gatherB(Xsm, XPITCH, tc - d, 0), acc);
      acc = wmma32(A1, gatherB(Xsm, XPITCH, tc - d, 32), acc);
      acc = wmma32(A2, gatherB(Xsm, XPITCH, tc, 0), acc);
      acc = wmma32(A3, gatherB(Xsm, XPITCH, tc, 32), acc);
      int t = t0 + ct * 16 + n;
      int tt = t / 75;  // cond repeat factor
#pragma unroll
      for (int r = 0; r < 8; ++r) {
        int row = rt * 16 + r + 8 * hs;
        float v = acc[r] + dil_b[l * 128 + row] +
                  cond256[((size_t)(b * NL + l) * 128 + row) * 256 + tt];
        Ysm[(ct * 16 + n) * YPITCH + row] = v;
      }
    }
  }
  __syncthreads();

  // ---- stage 2: gated activation ----
  for (int i = tid; i < 64 * 64; i += 256) {
    int t = i >> 6, c = i & 63;
    float y1 = Ysm[t * YPITCH + c];
    float y2 = Ysm[t * YPITCH + 64 + c];
    Gsm[t * GPITCH + c] = (h16)(tanhf(y1) * (1.f / (1.f + expf(-y2))));
  }
  __syncthreads();

  // ---- stage 3: skip GEMM (256x64) accumulated into global fp32 (L2-resident) ----
#pragma unroll
  for (int j = 0; j < 2; ++j) {
    int rt = wave * 2 + j;
    int base = (l * 16 + rt) * 2;
    v16h A0 = loadA(packW + OFF_SKP, base + 0);
    v16h A1 = loadA(packW + OFF_SKP, base + 1);
#pragma unroll
    for (int ct = 0; ct < 4; ++ct) {
      v8f acc = {0.f, 0.f, 0.f, 0.f, 0.f, 0.f, 0.f, 0.f};
      acc = wmma32(A0, gatherB(Gsm, GPITCH, ct * 16, 0), acc);
      acc = wmma32(A1, gatherB(Gsm, GPITCH, ct * 16, 32), acc);
      int t = t0 + ct * 16 + n;
      float* sp = skip + ((size_t)b * SLEN + t) * 256;
#pragma unroll
      for (int r = 0; r < 8; ++r) {
        int row = rt * 16 + r + 8 * hs;
        sp[row] += acc[r] + skp_b[l * 256 + row];
      }
    }
  }

  // ---- stage 4: residual GEMM (64x64) + x ----
  if (l < NL - 1) {
#pragma unroll
    for (int j = 0; j < 2; ++j) {
      int tile = wave * 2 + j;
      int rt = tile >> 2, ct = tile & 3;
      int base = (l * 4 + rt) * 2;
      v16h A0 = loadA(packW + OFF_RES, base + 0);
      v16h A1 = loadA(packW + OFF_RES, base + 1);
      v8f acc = {0.f, 0.f, 0.f, 0.f, 0.f, 0.f, 0.f, 0.f};
      acc = wmma32(A0, gatherB(Gsm, GPITCH, ct * 16, 0), acc);
      acc = wmma32(A1, gatherB(Gsm, GPITCH, ct * 16, 32), acc);
      int t = t0 + ct * 16 + n;
#pragma unroll
      for (int r = 0; r < 8; ++r) {
        int row = rt * 16 + r + 8 * hs;
        size_t o = ((size_t)b * SLEN + t) * 64 + row;
        xout[o] = (h16)(acc[r] + rs_b[l * 64 + row] + (float)xin[o]);
      }
    }
  }
}

// ---------------- output head ----------------
#define TH 32
#define SPITCH 258
#define LPITCH 260

__global__ __launch_bounds__(256) void head_kernel(const float* __restrict__ skip,
                                                   const h16* __restrict__ packW,
                                                   float* __restrict__ out) {
  // region A: Ssm (fp16 32x258 = 16512B) overlaid later by Lsm (fp32 32x260 = 33280B)
  // region B: H1sm (fp16 32x258 = 16512B)
  __shared__ __align__(16) unsigned char smraw[33280 + 16512];
  h16*   Ssm = (h16*)smraw;
  float* Lsm = (float*)smraw;
  h16*   H1  = (h16*)(smraw + 33280);
  const int tid = threadIdx.x, wave = tid >> 5, lane = tid & 31;
  const int b = blockIdx.x / (SLEN / TH);
  const int t0 = (blockIdx.x % (SLEN / TH)) * TH;
  const int n = lane & 15, hs = lane >> 4;

  for (int i = tid; i < TH * 256; i += 256) {
    int t = i >> 8, a = i & 255;
    float v = skip[((size_t)b * SLEN + t0 + t) * 256 + a];
    Ssm[t * SPITCH + a] = (h16)fmaxf(v, 0.f);
  }
  __syncthreads();

  // h1 = relu(out1_w @ relu(skip))
#pragma unroll
  for (int j = 0; j < 2; ++j) {
    int rt = wave * 2 + j;
    v16h A[8];
#pragma unroll
    for (int s = 0; s < 8; ++s) A[s] = loadA(packW + OFF_O1, rt * 8 + s);
#pragma unroll
    for (int ct = 0; ct < 2; ++ct) {
      v8f acc = {0.f, 0.f, 0.f, 0.f, 0.f, 0.f, 0.f, 0.f};
#pragma unroll
      for (int s = 0; s < 8; ++s)
        acc = wmma32(A[s], gatherB(Ssm, SPITCH, ct * 16, s * 32), acc);
#pragma unroll
      for (int r = 0; r < 8; ++r) {
        int row = rt * 16 + r + 8 * hs;
        H1[(ct * 16 + n) * SPITCH + row] = (h16)fmaxf(acc[r], 0.f);
      }
    }
  }
  __syncthreads();

  // logits = out2_w @ h1   (writes overlay region A; Ssm is dead)
#pragma unroll
  for (int j = 0; j < 2; ++j) {
    int rt = wave * 2 + j;
    v16h A[8];
#pragma unroll
    for (int s = 0; s < 8; ++s) A[s] = loadA(packW + OFF_O2, rt * 8 + s);
#pragma unroll
    for (int ct = 0; ct < 2; ++ct) {
      v8f acc = {0.f, 0.f, 0.f, 0.f, 0.f, 0.f, 0.f, 0.f};
#pragma unroll
      for (int s = 0; s < 8; ++s)
        acc = wmma32(A[s], gatherB(H1, SPITCH, ct * 16, s * 32), acc);
#pragma unroll
      for (int r = 0; r < 8; ++r) {
        int row = rt * 16 + r + 8 * hs;
        Lsm[(ct * 16 + n) * LPITCH + row] = acc[r];
      }
    }
  }
  __syncthreads();

  // log_softmax over channels; coalesced across t within fixed channel a
  if (tid < TH) {
    int t = tid;
    float mx = -1e30f;
    for (int a = 0; a < 256; ++a) mx = fmaxf(mx, Lsm[t * LPITCH + a]);
    float s = 0.f;
    for (int a = 0; a < 256; ++a) s += expf(Lsm[t * LPITCH + a] - mx);
    float lse = mx + logf(s);
    for (int a = 0; a < 256; ++a)
      out[((size_t)b * 256 + a) * SLEN + t0 + t] = Lsm[t * LPITCH + a] - lse;
  }
}

// ---------------- launch ----------------
extern "C" void kernel_launch(void* const* d_in, const int* in_sizes, int n_in,
                              void* d_out, int out_size, void* d_ws, size_t ws_size,
                              hipStream_t stream) {
  const float* lf   = (const float*)d_in[0];
  const float* gold = (const float*)d_in[1];
  const float* ew   = (const float*)d_in[2];
  const float* eb   = (const float*)d_in[3];
  const float* tw   = (const float*)d_in[4];
  const float* tb   = (const float*)d_in[5];
  const float* cw   = (const float*)d_in[6];
  const float* cb   = (const float*)d_in[7];
  const float* dw   = (const float*)d_in[8];
  const float* db   = (const float*)d_in[9];
  const float* rw   = (const float*)d_in[10];
  const float* rb   = (const float*)d_in[11];
  const float* sw   = (const float*)d_in[12];
  const float* sb   = (const float*)d_in[13];
  const float* o1   = (const float*)d_in[14];
  const float* o2   = (const float*)d_in[15];

  size_t off = 0;
  auto carve = [&](size_t bytes) {
    void* p = (char*)d_ws + off;
    off += (bytes + 255) & ~(size_t)255;
    return p;
  };
  h16*   packW  = (h16*)  carve((size_t)TOTAL_PACK * 2);
  h16*   xA     = (h16*)  carve((size_t)BN * SLEN * 64 * 2);
  h16*   xB     = (h16*)  carve((size_t)BN * SLEN * 64 * 2);
  float* up     = (float*)carve((size_t)BN * 80 * 256 * 4);
  float* cond   = (float*)carve((size_t)BN * NL * 128 * 256 * 4);
  float* skip   = (float*)carve((size_t)BN * SLEN * 256 * 4);

  pack_weights<<<(TOTAL_PACK + 255) / 256, 256, 0, stream>>>(dw, sw, rw, o1, o2, packW);
  up_kernel  <<<(BN * 80 * 256 + 255) / 256, 256, 0, stream>>>(lf, tw, tb, up);
  cond_kernel<<<(BN * NL * 128 * 256 + 255) / 256, 256, 0, stream>>>(cw, cb, up, cond);
  embed_kernel<<<(BN * SLEN * 64 + 255) / 256, 256, 0, stream>>>(gold, ew, eb, xA);
  int n4 = BN * SLEN * 256 / 4;
  zero_kernel<<<(n4 + 255) / 256, 256, 0, stream>>>((float4*)skip, n4);

  h16* xin = xA;
  h16* xout = xB;
  const int grid_layer = BN * (SLEN / 64);
  for (int l = 0; l < NL; ++l) {
    layer_kernel<<<grid_layer, 256, 0, stream>>>(l, 1 << (l % 6), xin, xout, packW,
                                                 cond, db, sb, rb, skip);
    h16* t = xin; xin = xout; xout = t;
  }
  head_kernel<<<BN * (SLEN / TH), 256, 0, stream>>>(skip, packW, (float*)d_out);
}
